// NormalBasedSDFLoss_618475291012
// MI455X (gfx1250) — compile-verified, matrix operand order checked by reading, not run on hardware
//
#include <hip/hip_runtime.h>
#include <hip/hip_bf16.h>
#include <math.h>

typedef __attribute__((ext_vector_type(16))) _Float16 v16h;
typedef __attribute__((ext_vector_type(8)))  _Float16 v8h;
typedef __attribute__((ext_vector_type(8)))  float    v8f;

#define HID 256
#define CHUNK 2048

__device__ __forceinline__ float softplus_f(float x) {
    if (x > 20.f)  return x;
    if (x < -20.f) return __expf(x);
    return log1pf(__expf(x));
}
__device__ __forceinline__ float sigmoid_f(float x) {
    return __builtin_amdgcn_rcpf(1.f + __expf(-x));   // v_rcp_f32, loss-accum precision
}

// ---------------------------------------------------------------------------
// f16 WMMA A/B fragment load (ISA 7.12.2 16-bit 16x32 layout):
// lane l holds row/col = l&15;
//   halves [0..7]  = K = kb + 8*(l>>4) + {0..7}   -> one 16B load
//   halves [8..15] = K = kb + 16 + 8*(l>>4) + {0..7} -> one 16B load
// src is row-major f16 with leading dim HID; both runs are 16B aligned.
// ---------------------------------------------------------------------------
__device__ __forceinline__ v16h load_frag_f16(const _Float16* __restrict__ src,
                                              int row, int kb, int lane) {
    const int hi = lane >> 4;
    const _Float16* r = src + (size_t)row * HID;
    union { v16h v; v8h h[2]; } u;
    u.h[0] = *(const v8h*)(r + kb + 8 * hi);
    u.h[1] = *(const v8h*)(r + kb + 16 + 8 * hi);
    return u.v;
}

// ---------------------------------------------------------------------------
// Init: y[m] = b3, grad = 0, accumulators = 0
// ---------------------------------------------------------------------------
__global__ void k_init(float* __restrict__ y, float* __restrict__ grad,
                       float* __restrict__ accum, const float* __restrict__ b3, int N) {
    int i = blockIdx.x * blockDim.x + threadIdx.x;
    if (i < 3 * N) grad[i] = 0.f;
    if (i < N)     y[i] = b3[0];
    if (i < 8)     accum[i] = 0.f;
}

// ---------------------------------------------------------------------------
// One-time weight staging: W2h = f16(W2) (row-major, serves W2^T operand),
// W2Th[n*HID+k] = f16(W2[k*HID+n]) (serves forward operand, contiguous k)
// ---------------------------------------------------------------------------
__global__ void k_prep_w2(const float* __restrict__ W2, _Float16* __restrict__ W2h,
                          _Float16* __restrict__ W2Th) {
    int idx = blockIdx.x * blockDim.x + threadIdx.x;
    if (idx >= HID * HID) return;
    int k = idx >> 8, n = idx & 255;
    float w = W2[idx];
    W2h[idx] = (_Float16)w;
    W2Th[n * HID + k] = (_Float16)w;
}

// ---------------------------------------------------------------------------
// Layer 1: h1 = softplus(P @ W1 + b1), K=3 -> plain VALU; emit f16 for WMMA
// ---------------------------------------------------------------------------
__global__ void k_layer1(const float* __restrict__ P, const float* __restrict__ W1,
                         const float* __restrict__ b1, _Float16* __restrict__ h1h, int N) {
    int idx = blockIdx.x * blockDim.x + threadIdx.x;
    if (idx >= N * HID) return;
    int n = idx >> 8, j = idx & 255;
    float z = b1[j];
    z = fmaf(P[n * 3 + 0], W1[j],           z);
    z = fmaf(P[n * 3 + 1], W1[HID + j],     z);
    z = fmaf(P[n * 3 + 2], W1[2 * HID + j], z);
    h1h[idx] = (_Float16)softplus_f(z);
}

// ---------------------------------------------------------------------------
// Layer 2 forward (WMMA) fused with layer-3 contraction and backward seed:
//   z2 = h1 @ W2 + b2 ; y[m] += sum_n softplus(z2)*W3[n] ; g2 = sigmoid(z2)*W3[n]
// one wave per 16x64 strip; f16 operands, f32 accumulate
// ---------------------------------------------------------------------------
__global__ void k_fwd2(const _Float16* __restrict__ h1h, const _Float16* __restrict__ W2Th,
                       const float* __restrict__ b2, const float* __restrict__ W3,
                       _Float16* __restrict__ g2h, float* __restrict__ y, int N) {
    const int lane = threadIdx.x & 31;
    const int hi = lane >> 4, nl = lane & 15;
    const int mBase = blockIdx.x * 16;
    const int nGroup = blockIdx.y * 64;
    const int row = mBase + nl;

    v8f acc[4] = {};
    for (int kb = 0; kb < HID; kb += 32) {
        v16h a = load_frag_f16(h1h, row, kb, lane);
        #pragma unroll
        for (int t = 0; t < 4; ++t) {
            v16h b = load_frag_f16(W2Th, nGroup + 16 * t + nl, kb, lane);
            acc[t] = __builtin_amdgcn_wmma_f32_16x16x32_f16(
                false, a, false, b, (short)0, acc[t], false, false);
        }
    }
    #pragma unroll
    for (int t = 0; t < 4; ++t) {
        const int n = nGroup + 16 * t + nl;
        const float w3 = W3[n];
        const float bias = b2[n];
        #pragma unroll
        for (int r = 0; r < 8; ++r) {
            const int m = mBase + r + 8 * hi;   // C/D layout: vgpr r, lane-half hi
            const float z = acc[t][r] + bias;
            g2h[(size_t)m * HID + n] = (_Float16)(sigmoid_f(z) * w3);
            float yp = softplus_f(z) * w3;
            yp += __shfl_xor(yp, 1, 16);
            yp += __shfl_xor(yp, 2, 16);
            yp += __shfl_xor(yp, 4, 16);
            yp += __shfl_xor(yp, 8, 16);
            if (nl == 0) atomicAdd(&y[m], yp);
        }
    }
}

// ---------------------------------------------------------------------------
// Backward (WMMA): g1 = (g2 @ W2^T) * sigmoid(z1), grad = g1 @ W1^T.
// z1 recomputed from P (K=3, cheap). W2^T operand = W2h rows (contiguous k).
// ---------------------------------------------------------------------------
__global__ void k_bwd(const _Float16* __restrict__ g2h, const _Float16* __restrict__ W2h,
                      const float* __restrict__ P, const float* __restrict__ W1,
                      const float* __restrict__ b1, float* __restrict__ grad, int N) {
    const int lane = threadIdx.x & 31;
    const int hi = lane >> 4, nl = lane & 15;
    const int mBase = blockIdx.x * 16;
    const int nGroup = blockIdx.y * 64;
    const int row = mBase + nl;

    v8f acc[4] = {};
    for (int kb = 0; kb < HID; kb += 32) {
        v16h a = load_frag_f16(g2h, row, kb, lane);
        #pragma unroll
        for (int t = 0; t < 4; ++t) {
            v16h b = load_frag_f16(W2h, nGroup + 16 * t + nl, kb, lane);
            acc[t] = __builtin_amdgcn_wmma_f32_16x16x32_f16(
                false, a, false, b, (short)0, acc[t], false, false);
        }
    }
    #pragma unroll
    for (int t = 0; t < 4; ++t) {
        const int n = nGroup + 16 * t + nl;
        const float w10 = W1[n], w11 = W1[HID + n], w12 = W1[2 * HID + n];
        const float bias = b1[n];
        #pragma unroll
        for (int r = 0; r < 8; ++r) {
            const int m = mBase + r + 8 * hi;
            float z1 = bias;
            z1 = fmaf(P[m * 3 + 0], w10, z1);
            z1 = fmaf(P[m * 3 + 1], w11, z1);
            z1 = fmaf(P[m * 3 + 2], w12, z1);
            const float g1 = acc[t][r] * sigmoid_f(z1);
            float p0 = g1 * w10, p1 = g1 * w11, p2 = g1 * w12;
            p0 += __shfl_xor(p0, 1, 16); p0 += __shfl_xor(p0, 2, 16);
            p0 += __shfl_xor(p0, 4, 16); p0 += __shfl_xor(p0, 8, 16);
            p1 += __shfl_xor(p1, 1, 16); p1 += __shfl_xor(p1, 2, 16);
            p1 += __shfl_xor(p1, 4, 16); p1 += __shfl_xor(p1, 8, 16);
            p2 += __shfl_xor(p2, 1, 16); p2 += __shfl_xor(p2, 2, 16);
            p2 += __shfl_xor(p2, 4, 16); p2 += __shfl_xor(p2, 8, 16);
            if (nl == 0) {
                atomicAdd(&grad[m * 3 + 0], p0);
                atomicAdd(&grad[m * 3 + 1], p1);
                atomicAdd(&grad[m * 3 + 2], p2);
            }
        }
    }
}

// ---------------------------------------------------------------------------
// Surface losses: sdf (y^2), eikonal ((|g|-1)^2), grad-normal (|g-n|^2)
// ---------------------------------------------------------------------------
__global__ void k_surface_loss(const float* __restrict__ y, const float* __restrict__ grad,
                               const float* __restrict__ nrm, float* __restrict__ accum, int N) {
    int m = blockIdx.x * blockDim.x + threadIdx.x;
    float s0 = 0.f, s1 = 0.f, s2 = 0.f;
    if (m < N) {
        float yv = y[m];
        float gx = grad[m * 3], gy = grad[m * 3 + 1], gz = grad[m * 3 + 2];
        float gn = sqrtf(gx * gx + gy * gy + gz * gz) - 1.f;
        float nx = nrm[m * 3], ny = nrm[m * 3 + 1], nz = nrm[m * 3 + 2];
        float dx = gx - nx, dy = gy - ny, dz = gz - nz;
        s0 = yv * yv;
        s1 = gn * gn;
        s2 = dx * dx + dy * dy + dz * dz;
    }
    #pragma unroll
    for (int off = 1; off < 32; off <<= 1) {
        s0 += __shfl_xor(s0, off, 32);
        s1 += __shfl_xor(s1, off, 32);
        s2 += __shfl_xor(s2, off, 32);
    }
    if ((threadIdx.x & 31) == 0) {
        atomicAdd(&accum[0], s0);
        atomicAdd(&accum[1], s1);
        atomicAdd(&accum[2], s2);
    }
}

// ---------------------------------------------------------------------------
// 2-NN orientation loss: one thread per query, surface points streamed through
// LDS in 24KB chunks (broadcast reads, no bank conflicts). Running top-2.
// ---------------------------------------------------------------------------
__global__ void k_orientation(const float* __restrict__ query, const float* __restrict__ surf,
                              const float* __restrict__ nrm, const float* __restrict__ pred,
                              float* __restrict__ accum, int accIdx, int M, int N) {
    __shared__ float sp[CHUNK * 3];
    const int tid = threadIdx.x;
    const int m = blockIdx.x * blockDim.x + tid;
    float qx = 0.f, qy = 0.f, qz = 0.f;
    if (m < M) { qx = query[m * 3]; qy = query[m * 3 + 1]; qz = query[m * 3 + 2]; }
    float d0 = 3.4e38f, d1 = 3.4e38f;
    int i0 = 0, i1 = 0;
    for (int base = 0; base < N; base += CHUNK) {
        const int cnt = min(CHUNK, N - base);
        __syncthreads();
        for (int t = tid; t < cnt * 3; t += blockDim.x) sp[t] = surf[(size_t)base * 3 + t];
        __syncthreads();
        if (m < M) {
            for (int j = 0; j < cnt; ++j) {
                float dx = qx - sp[j * 3], dy = qy - sp[j * 3 + 1], dz = qz - sp[j * 3 + 2];
                float d = fmaf(dx, dx, fmaf(dy, dy, dz * dz));
                int gi = base + j;
                if (d < d0)      { d1 = d0; i1 = i0; d0 = d; i0 = gi; }
                else if (d < d1) { d1 = d; i1 = gi; }
            }
        }
    }
    float loss = 0.f;
    if (m < M) {
        float dots = 0.f;
        int idxs[2] = { i0, i1 };
        #pragma unroll
        for (int j = 0; j < 2; ++j) {
            int i = idxs[j];
            float vx = qx - surf[i * 3], vy = qy - surf[i * 3 + 1], vz = qz - surf[i * 3 + 2];
            dots += vx * nrm[i * 3] + vy * nrm[i * 3 + 1] + vz * nrm[i * 3 + 2];
        }
        float mean = 0.5f * dots;
        float sgn = (mean > 0.f) ? 1.f : ((mean < 0.f) ? -1.f : 0.f);
        loss = fmaxf(0.f, -pred[m] * sgn);
    }
    #pragma unroll
    for (int off = 1; off < 32; off <<= 1) loss += __shfl_xor(loss, off, 32);
    if ((tid & 31) == 0) atomicAdd(&accum[accIdx], loss);
}

// ---------------------------------------------------------------------------
// Finalize: means + weighted total -> 6 outputs
// ---------------------------------------------------------------------------
__global__ void k_finalize(const float* __restrict__ accum, float* __restrict__ out,
                           int N, int M) {
    if (threadIdx.x != 0 || blockIdx.x != 0) return;
    float sdf   = accum[0] / (float)N;
    float eik   = accum[1] / (float)N;
    float gradn = accum[2] / (3.f * (float)N);
    float ori   = accum[3] / (float)M;
    float nori  = accum[4] / (float)M;
    float total = 7000.f * sdf + 600.f * eik + 500.f * ori + 10.f * nori + 200.f * gradn;
    out[0] = total; out[1] = sdf; out[2] = eik; out[3] = ori; out[4] = nori; out[5] = gradn;
}

// ---------------------------------------------------------------------------
extern "C" void kernel_launch(void* const* d_in, const int* in_sizes, int n_in,
                              void* d_out, int out_size, void* d_ws, size_t ws_size,
                              hipStream_t stream) {
    const float* surf   = (const float*)d_in[0];
    const float* nrm    = (const float*)d_in[1];
    const float* offp   = (const float*)d_in[2];
    const float* nearp  = (const float*)d_in[3];
    const float* nmpred = (const float*)d_in[4];
    const float* nppred = (const float*)d_in[5];
    const float* W1 = (const float*)d_in[6];
    const float* b1 = (const float*)d_in[7];
    const float* W2 = (const float*)d_in[8];
    const float* b2 = (const float*)d_in[9];
    const float* W3 = (const float*)d_in[10];
    const float* b3 = (const float*)d_in[11];
    const int N = in_sizes[0] / 3;   // 8192
    const int M = in_sizes[2] / 3;   // 32768

    // workspace layout (16B-aligned slabs)
    char* wsb = (char*)d_ws;
    _Float16* h1h  = (_Float16*)wsb;                               // [N,256] f16
    _Float16* g2h  = (_Float16*)(wsb + (size_t)N * HID * 2);       // [N,256] f16
    _Float16* W2h  = (_Float16*)(wsb + (size_t)N * HID * 4);       // [256,256] f16
    _Float16* W2Th = W2h + (size_t)HID * HID;                      // [256,256] f16
    float* y     = (float*)(W2Th + (size_t)HID * HID);             // [N]
    float* grad  = y + N;                                          // [N,3]
    float* accum = grad + (size_t)N * 3;                           // [8]

    k_init<<<dim3((3 * N + 255) / 256), dim3(256), 0, stream>>>(y, grad, accum, b3, N);
    k_prep_w2<<<dim3((HID * HID + 255) / 256), dim3(256), 0, stream>>>(W2, W2h, W2Th);
    k_layer1<<<dim3((N * HID + 255) / 256), dim3(256), 0, stream>>>(surf, W1, b1, h1h, N);

    dim3 gemmGrid(N / 16, HID / 64);   // (512, 4) waves of 32
    k_fwd2<<<gemmGrid, dim3(32), 0, stream>>>(h1h, W2Th, b2, W3, g2h, y, N);
    k_bwd <<<gemmGrid, dim3(32), 0, stream>>>(g2h, W2h, surf, W1, b1, grad, N);

    k_surface_loss<<<dim3((N + 255) / 256), dim3(256), 0, stream>>>(y, grad, nrm, accum, N);
    k_orientation<<<dim3((M + 255) / 256), dim3(256), 0, stream>>>(offp,  surf, nrm, nmpred, accum, 3, M, N);
    k_orientation<<<dim3((M + 255) / 256), dim3(256), 0, stream>>>(nearp, surf, nrm, nppred, accum, 4, M, N);
    k_finalize<<<dim3(1), dim3(1), 0, stream>>>(accum, (float*)d_out, N, M);
}
